// toEdges_65197603553904
// MI455X (gfx1250) — matrix operand order repaired and checked by reading, not compiled, not run on hardware
//
#include <hip/hip_runtime.h>

// MI455X / gfx1250, wave32. bf16 WMMA path: v_wmma_f32_16x16x32_bf16.
// Round 3: async global->LDS staging (GLOBAL_LOAD_ASYNC_TO_LDS_B128 / ASYNCcnt)
// using the toolchain's actual builtin signature: (int4 AS1*, int4 AS3*, Ii, Ii).

typedef __attribute__((ext_vector_type(16))) __bf16 v16bf;
typedef __attribute__((ext_vector_type(8)))  __bf16 v8bf;
typedef __attribute__((ext_vector_type(4)))  __bf16 v4bf;
typedef __attribute__((ext_vector_type(8)))  float  v8f;

#define BB 8
#define NN 1024
#define XX 64
#define FF 8
#define PAD 72   // LDS row stride in bf16 elements (144B) -> bank-conflict friendly, 16B aligned

#if defined(__has_builtin)
#  if __has_builtin(__builtin_amdgcn_global_load_async_to_lds_b128)
#    define HAVE_ASYNC_LDS 1
#  endif
#endif
#ifndef HAVE_ASYNC_LDS
#  define HAVE_ASYNC_LDS 0
#endif

#if HAVE_ASYNC_LDS
// Match the diagnostic's parameter type: gcc-style vector_size(16) int,
// address-space qualified (HIP prints AS1 as __device__, AS3 as __shared__).
typedef int v4i_n __attribute__((vector_size(16)));
typedef __attribute__((address_space(1))) v4i_n* gv4i_p;
typedef __attribute__((address_space(3))) v4i_n* lv4i_p;

// 16B global -> LDS direct copy, tracked by ASYNCcnt. Pointers rebuilt via
// integer round-trips: AS1 keeps the full 64-bit address; AS3 is the low 32
// bits of the generic pointer (== LDS byte offset per the aperture mapping).
__device__ __forceinline__ void cp16_async(const void* gsrc, void* ldst) {
  __builtin_amdgcn_global_load_async_to_lds_b128(
      (gv4i_p)(unsigned long long)gsrc,
      (lv4i_p)(unsigned)(unsigned long long)ldst,
      0, 0);
}

__device__ __forceinline__ void wait_async0() {
#if __has_builtin(__builtin_amdgcn_s_wait_asynccnt)
  __builtin_amdgcn_s_wait_asynccnt(0);
#else
  asm volatile("s_wait_asynccnt 0x0" ::: "memory");
#endif
}
#endif  // HAVE_ASYNC_LDS

__device__ __forceinline__ __bf16 f2bf(float f) {
  union { float f; unsigned u; } c; c.f = f;
  unsigned r = (c.u + 0x7FFFu + ((c.u >> 16) & 1u)) >> 16;  // round-to-nearest-even
  union { unsigned short s; __bf16 b; } o; o.s = (unsigned short)r;
  return o.b;
}

// ---------------- kernel 0: V f32 -> bf16 ----------------
__global__ __launch_bounds__(256) void k_cvt_v(const float* __restrict__ V,
                                               __bf16* __restrict__ Vbf) {
  int i = (blockIdx.x * 256 + threadIdx.x) * 4;
  const float4 v = *(const float4*)(V + i);
  v4bf o; o[0] = f2bf(v.x); o[1] = f2bf(v.y); o[2] = f2bf(v.z); o[3] = f2bf(v.w);
  *(v4bf*)(Vbf + i) = o;
}

// ---------------- kernel 1: aux[b][f][u][y] = V[b] @ W[:,:,f]  (bf16 out) ----------------
__global__ __launch_bounds__(256) void k_aux(const float* __restrict__ V,
                                             const float* __restrict__ W,
                                             __bf16* __restrict__ aux) {
  __shared__ __align__(16) __bf16 sVu[128 * PAD];  // u-tile rows, y-major
  __shared__ __align__(16) __bf16 sW[64 * PAD];    // TRANSPOSED: sW[y*PAD + x]
  const int b = blockIdx.z, f = blockIdx.y, u0 = blockIdx.x * 128;
  const int t = threadIdx.x;

  // stage V u-tile (128x64 f32 -> bf16)  [conversion -> must go through VGPRs]
#pragma unroll
  for (int c = 0; c < 8; ++c) {
    int cid = c * 256 + t;               // 2048 chunks of 4 floats
    int row = cid >> 4, q = cid & 15;
    const float4 v = *(const float4*)(V + ((size_t)b * NN + u0 + row) * XX + q * 4);
    v4bf o; o[0] = f2bf(v.x); o[1] = f2bf(v.y); o[2] = f2bf(v.z); o[3] = f2bf(v.w);
    *(v4bf*)(sVu + row * PAD + q * 4) = o;
  }
  // stage W[:,:,f] transposed (strided gather, tiny + L2 resident)
#pragma unroll
  for (int c = 0; c < 16; ++c) {
    int e = c * 256 + t;                 // 4096 elements
    int x = e >> 6, y = e & 63;
    sW[y * PAD + x] = f2bf(W[((size_t)x * XX + y) * FF + f]);
  }
  __syncthreads();

  const int w = t >> 5, lane = t & 31, hi = lane >> 4, col = lane & 15;

  // A fragments (16x32 bf16), one per K-step, shared across N tiles
  v16bf A[2];
#pragma unroll
  for (int ks = 0; ks < 2; ++ks) {
    int base = (w * 16 + col) * PAD + ks * 32 + hi * 8;
    v8bf lo = *(const v8bf*)(sVu + base);
    v8bf hh = *(const v8bf*)(sVu + base + 16);
#pragma unroll
    for (int e = 0; e < 8; ++e) { A[ks][e] = lo[e]; A[ks][8 + e] = hh[e]; }
  }

#pragma unroll
  for (int nt = 0; nt < 4; ++nt) {
    v8f acc = {};
#pragma unroll
    for (int ks = 0; ks < 2; ++ks) {
      int base = (nt * 16 + col) * PAD + ks * 32 + hi * 8;
      v8bf lo = *(const v8bf*)(sW + base);
      v8bf hh = *(const v8bf*)(sW + base + 16);
      v16bf Bf;
#pragma unroll
      for (int e = 0; e < 8; ++e) { Bf[e] = lo[e]; Bf[8 + e] = hh[e]; }
      acc = __builtin_amdgcn_wmma_f32_16x16x32_bf16(false, A[ks], false, Bf,
                                                    (short)0, acc, false, false);
    }
#pragma unroll
    for (int r = 0; r < 8; ++r) {
      int u = u0 + w * 16 + r + hi * 8;
      int y = nt * 16 + col;
      aux[(((size_t)b * FF + f) * NN + u) * XX + y] = f2bf(acc[r]);
    }
  }
}

// ---------------- kernel 2: E[b,u,v,f] = relu(aux[b,f,u,:] . Vbf[b,v,:] + bias[f]) ----------------
__global__ __launch_bounds__(256) void k_main(const __bf16* __restrict__ aux,
                                              const __bf16* __restrict__ Vbf,
                                              const float* __restrict__ bias,
                                              float* __restrict__ out) {
  __shared__ __align__(16) __bf16 sAux[FF * 32 * PAD];  // [f][urow][y]
  __shared__ __align__(16) __bf16 sVv[64 * PAD];        // [vrow][y]
  const int b = blockIdx.z;
  const int u0 = blockIdx.x * 32;
  const int v0 = blockIdx.y * 64;
  const int t = threadIdx.x;

#if HAVE_ASYNC_LDS
  // stage aux tiles (8f x 32 rows x 32 halves = 1024 x 16B) via async LDS DMA
#pragma unroll
  for (int c = 0; c < 4; ++c) {
    int cid = c * 256 + t;
    int f = cid >> 7, rem = cid & 127, row = rem >> 2, q = rem & 3;
    cp16_async(aux + (((size_t)b * FF + f) * NN + u0 + row) * XX + q * 8,
               sAux + (f * 32 + row) * PAD + q * 8);
  }
  // stage Vv tile (64 rows x 32 halves = 256 x 16B)
  {
    int row = t >> 2, q = t & 3;
    cp16_async(Vbf + ((size_t)b * NN + v0 + row) * XX + q * 8,
               sVv + row * PAD + q * 8);
  }
  wait_async0();
#else
#pragma unroll
  for (int c = 0; c < 4; ++c) {
    int cid = c * 256 + t;
    int f = cid >> 7, rem = cid & 127, row = rem >> 2, q = rem & 3;
    const v8bf g = *(const v8bf*)(aux + (((size_t)b * FF + f) * NN + u0 + row) * XX + q * 8);
    *(v8bf*)(sAux + (f * 32 + row) * PAD + q * 8) = g;
  }
  {
    int row = t >> 2, q = t & 3;
    const v8bf g = *(const v8bf*)(Vbf + ((size_t)b * NN + v0 + row) * XX + q * 8);
    *(v8bf*)(sVv + row * PAD + q * 8) = g;
  }
#endif
  __syncthreads();

  const int w = t >> 5, lane = t & 31, hi = lane >> 4, col = lane & 15;
  const int m = w >> 2, n = w & 3;  // wave's 16x16 (u,v) sub-tile

  // B fragments (Vv^T, K=y), shared across all 8 features
  v16bf Bf[2];
#pragma unroll
  for (int ks = 0; ks < 2; ++ks) {
    int base = (n * 16 + col) * PAD + ks * 32 + hi * 8;
    v8bf lo = *(const v8bf*)(sVv + base);
    v8bf hh = *(const v8bf*)(sVv + base + 16);
#pragma unroll
    for (int e = 0; e < 8; ++e) { Bf[ks][e] = lo[e]; Bf[ks][8 + e] = hh[e]; }
  }

  v8f acc[FF];
#pragma unroll
  for (int f = 0; f < FF; ++f) {
    v8f a = {};
#pragma unroll
    for (int ks = 0; ks < 2; ++ks) {
      int base = (f * 32 + m * 16 + col) * PAD + ks * 32 + hi * 8;
      v8bf lo = *(const v8bf*)(sAux + base);
      v8bf hh = *(const v8bf*)(sAux + base + 16);
      v16bf Af;
#pragma unroll
      for (int e = 0; e < 8; ++e) { Af[e] = lo[e]; Af[8 + e] = hh[e]; }
      a = __builtin_amdgcn_wmma_f32_16x16x32_bf16(false, Af, false, Bf[ks],
                                                  (short)0, a, false, false);
    }
    acc[f] = a;
  }

  float bv[FF];
#pragma unroll
  for (int f = 0; f < FF; ++f) bv[f] = bias[f];

  // epilogue: each lane owns all 8 f-values of its (u,v) pairs -> 32B contiguous stores
#pragma unroll
  for (int r = 0; r < 8; ++r) {
    int u = u0 + m * 16 + r + hi * 8;
    int v = v0 + n * 16 + col;
    float4 lo4, hi4;
    lo4.x = fmaxf(acc[0][r] + bv[0], 0.f);
    lo4.y = fmaxf(acc[1][r] + bv[1], 0.f);
    lo4.z = fmaxf(acc[2][r] + bv[2], 0.f);
    lo4.w = fmaxf(acc[3][r] + bv[3], 0.f);
    hi4.x = fmaxf(acc[4][r] + bv[4], 0.f);
    hi4.y = fmaxf(acc[5][r] + bv[5], 0.f);
    hi4.z = fmaxf(acc[6][r] + bv[6], 0.f);
    hi4.w = fmaxf(acc[7][r] + bv[7], 0.f);
    float* p = out + (((size_t)b * NN + u) * NN + v) * FF;
    *(float4*)(p)     = lo4;
    *(float4*)(p + 4) = hi4;
  }
}

extern "C" void kernel_launch(void* const* d_in, const int* in_sizes, int n_in,
                              void* d_out, int out_size, void* d_ws, size_t ws_size,
                              hipStream_t stream) {
  (void)in_sizes; (void)n_in; (void)out_size; (void)ws_size;
  const float* V    = (const float*)d_in[0];
  const float* W    = (const float*)d_in[1];
  const float* bias = (const float*)d_in[2];
  float* out = (float*)d_out;

  __bf16* aux = (__bf16*)d_ws;                                          // 8 MB
  __bf16* vbf = (__bf16*)((char*)d_ws + (size_t)BB * FF * NN * XX * 2); // +1 MB

  k_cvt_v<<<dim3((BB * NN * XX) / 1024), 256, 0, stream>>>(V, vbf);
  k_aux <<<dim3(NN / 128, FF, BB), 256, 0, stream>>>(V, W, aux);
  k_main<<<dim3(NN / 32, NN / 64, BB), 256, 0, stream>>>(aux, vbf, bias, out);
}
